// MultiHeadSelfAttention_55774445306649
// MI455X (gfx1250) — matrix operand order repaired and checked by reading, not compiled, not run on hardware
//
#include <hip/hip_runtime.h>
#include <math.h>
#include <stdint.h>

// ---------------- problem constants (B=4, T=2048, C=1024, H=16, D=64) -------
constexpr int B_ = 4;
constexpr int T_ = 2048;
constexpr int C_ = 1024;
constexpr int H_ = 16;
constexpr int D_ = 64;
constexpr float NEG_INF_ = -1000000000.0f;

typedef __attribute__((ext_vector_type(16))) __bf16 v16bf;
typedef __attribute__((ext_vector_type(8)))  __bf16 v8bf;
typedef __attribute__((ext_vector_type(8)))  float  v8f;
typedef __attribute__((ext_vector_type(4)))  unsigned int u32x4;
typedef __attribute__((ext_vector_type(8)))  int i32x8;
typedef __attribute__((ext_vector_type(4)))  int i32x4;

union BF16Frag {
    v16bf v;
    v8bf  h[2];
};

__device__ __forceinline__ v8f wmma_bf16(const BF16Frag& a, const BF16Frag& b, v8f c) {
    // D = A(16x32 bf16) x B(32x16 bf16) + C(16x16 f32)
    return __builtin_amdgcn_wmma_f32_16x16x32_bf16(
        /*neg_a=*/false, a.v, /*neg_b=*/false, b.v,
        /*c_mod=*/(short)0, c, /*reuse_a=*/false, /*reuse_b=*/false);
}

__device__ __forceinline__ v8f vzero8() {
    v8f z = {0.f, 0.f, 0.f, 0.f, 0.f, 0.f, 0.f, 0.f};
    return z;
}

// ---- Tensor Data Mover: 2D tile global->LDS with LDS row padding -----------
// D# layout per CDNA5 ISA 8.3/8.4.  data_size=2 bytes.  pad_interval code k
// means pad after 2^(k+1) DWORDs; pad_amount code a means (a+1) DWORDs pad.
// This toolchain exposes the 6-arg builtin:
//   (uint32x4 g0, int32x8 g1, int32x4 g2, int32x4 g3, int32x8, i32 cpol)
__device__ __forceinline__ void tdm_load_tile_2d(
    unsigned lds_off, const void* gptr,
    unsigned tensor_d0, unsigned tensor_d1,
    unsigned tile_d0, unsigned tile_d1,
    unsigned long long d0_stride,        // elements
    unsigned pad_interval, unsigned pad_amount) {

    unsigned long long ga = (unsigned long long)gptr;
    u32x4 g0;
    g0[0] = 1u;                                   // count=1 (valid user D#)
    g0[1] = lds_off;                              // lds_addr (bytes)
    g0[2] = (unsigned)(ga & 0xffffffffull);       // global_addr[31:0]
    g0[3] = (unsigned)(ga >> 32) | (2u << 30);    // global_addr[56:32] | type=2

    i32x8 g1;
    g1[0] = (int)((1u << 16)                      // data_size = 2 bytes
                | (1u << 20)                      // pad_enable
                | (pad_interval << 22)
                | (pad_amount << 25));
    g1[1] = (int)((tensor_d0 & 0xffffu) << 16);                     // tdim0 lo
    g1[2] = (int)(((tensor_d0 >> 16) & 0xffffu) |
                  ((tensor_d1 & 0xffffu) << 16));                   // tdim0 hi | tdim1 lo
    g1[3] = (int)(((tensor_d1 >> 16) & 0xffffu) |
                  ((tile_d0 & 0xffffu) << 16));                     // tdim1 hi | tile0
    g1[4] = (int)(tile_d1 & 0xffffu);                               // tile1 (tile2=0)
    g1[5] = (int)(unsigned)(d0_stride & 0xffffffffull);             // stride0 lo
    g1[6] = (int)(unsigned)((d0_stride >> 32) & 0xffffu);           // stride0 hi
    g1[7] = 0;

    i32x4 gz4 = {0, 0, 0, 0};
    i32x8 gz8 = {0, 0, 0, 0, 0, 0, 0, 0};
    __builtin_amdgcn_tensor_load_to_lds(g0, g1, gz4, gz4, gz8, 0);
}

// ---------------- stage 0: f32 -> bf16 convert / transposed weight convert --
__global__ __launch_bounds__(256) void convert_f32_bf16(const float* __restrict__ src,
                                                        __bf16* __restrict__ dst, int n) {
    int i = blockIdx.x * 256 + threadIdx.x;
    if (i < n) dst[i] = (__bf16)src[i];
}

// w[K][N] (f32) -> wT[N][K] (bf16)
__global__ __launch_bounds__(256) void transpose_to_bf16(const float* __restrict__ w,
                                                         __bf16* __restrict__ wT,
                                                         int K, int N) {
    int i = blockIdx.x * 256 + threadIdx.x;
    if (i < K * N) {
        int n = i / K;
        int k = i - n * K;
        wT[i] = (__bf16)w[(long)k * N + n];
    }
}

// ---------------- stage 1/3: tiled WMMA GEMM --------------------------------
// C[M][N] = A[M][K] x Bt[N][K]^T.  Block tile 128x128, K-step 32, 8 waves,
// each wave computes a 32(M) x 64(N) sub-tile = 2x4 WMMA accumulators.
// A/B tiles staged to LDS by the Tensor Data Mover (wave 0 issues, waits
// tensorcnt, block barrier publishes).
// MODE 0: QKV epilogue -> scatter bf16 into q (scaled), k, vT buffers.
// MODE 1: plain f32 store to outf.
template <int MODE>
__global__ __launch_bounds__(256) void gemm_bf16_wmma(
    const __bf16* __restrict__ A, const __bf16* __restrict__ Bt,
    int M, int N, int K,
    float* __restrict__ outf,
    __bf16* __restrict__ qb, __bf16* __restrict__ kb, __bf16* __restrict__ vTb) {

    constexpr int LDT = 40;  // 32 elem row + 8 elem pad (TDM: interval 3, amount 3)
    __shared__ alignas(16) __bf16 a_lds[128 * LDT];
    __shared__ alignas(16) __bf16 b_lds[128 * LDT];

    const int tid  = threadIdx.x;
    const int lane = tid & 31;
    const int wave = tid >> 5;
    const int bm = blockIdx.x * 128;
    const int bn = blockIdx.y * 128;
    const int wm = (wave & 3) * 32;   // wave M offset within block tile
    const int wn = (wave >> 2) * 64;  // wave N offset within block tile

    const int mrow   = lane & 15;         // row/col within 16 for frag loads
    const int akb    = (lane >> 4) * 16;  // A-operand k base (contiguous 16)
    const int bkb    = (lane >> 4) * 16;  // B-operand k base (contiguous 16)
    const int rowadd = (lane >> 4) * 8;   // C/D layout row offset

    const unsigned a_off = (unsigned)(uintptr_t)(void*)a_lds;
    const unsigned b_off = (unsigned)(uintptr_t)(void*)b_lds;

    v8f acc[2][4];
#pragma unroll
    for (int i = 0; i < 2; ++i)
#pragma unroll
        for (int j = 0; j < 4; ++j) acc[i][j] = vzero8();

    for (int kk = 0; kk < K; kk += 32) {
        __syncthreads();  // previous tile fully consumed
        if (wave == 0) {
            tdm_load_tile_2d(a_off, A + (long)bm * K + kk,
                             32, 128, 32, 128, (unsigned long long)K, 3, 3);
            tdm_load_tile_2d(b_off, Bt + (long)bn * K + kk,
                             32, 128, 32, 128, (unsigned long long)K, 3, 3);
            __builtin_amdgcn_s_wait_tensorcnt(0);
        }
        __syncthreads();  // tile visible to all waves

        BF16Frag af[2];
#pragma unroll
        for (int mi = 0; mi < 2; ++mi) {
            int r = wm + mi * 16 + mrow;
            af[mi].h[0] = *(const v8bf*)&a_lds[r * LDT + akb];
            af[mi].h[1] = *(const v8bf*)&a_lds[r * LDT + akb + 8];
        }
#pragma unroll
        for (int ni = 0; ni < 4; ++ni) {
            BF16Frag bf;
            int r = wn + ni * 16 + mrow;
            bf.h[0] = *(const v8bf*)&b_lds[r * LDT + bkb];
            bf.h[1] = *(const v8bf*)&b_lds[r * LDT + bkb + 8];
#pragma unroll
            for (int mi = 0; mi < 2; ++mi)
                acc[mi][ni] = wmma_bf16(af[mi], bf, acc[mi][ni]);
        }
    }

    // epilogue
#pragma unroll
    for (int mi = 0; mi < 2; ++mi)
#pragma unroll
        for (int ni = 0; ni < 4; ++ni)
#pragma unroll
            for (int r = 0; r < 8; ++r) {
                int m = bm + wm + mi * 16 + r + rowadd;
                int n = bn + wn + ni * 16 + mrow;
                float val = acc[mi][ni][r];
                if (MODE == 1) {
                    outf[(long)m * N + n] = val;
                } else {
                    // n = s*1024 + head*64 + d ; m = b*T + t
                    int s3  = n >> 10;
                    int rem = n & 1023;
                    int hh  = rem >> 6;
                    int dd  = rem & 63;
                    int bb  = m >> 11;          // T = 2048
                    int tt  = m & (T_ - 1);
                    long bhb = (long)bb * H_ + hh;
                    if (s3 == 0)
                        qb[(bhb * T_ + tt) * D_ + dd] = (__bf16)(val * 0.125f); // D^-0.5
                    else if (s3 == 1)
                        kb[(bhb * T_ + tt) * D_ + dd] = (__bf16)val;
                    else
                        vTb[(bhb * D_ + dd) * T_ + tt] = (__bf16)val;           // V^T
                }
            }
}

// ---------------- stage 2: flash attention ----------------------------------
// One block per (b, h, 128-query slab). 8 waves, 16 queries per wave.
// K tile [32][64] and V^T tile [64][32] staged via the Tensor Data Mover;
// online softmax per wave; P restaged through per-wave LDS to A-layout.
__global__ __launch_bounds__(256) void flash_attn_wmma(
    const __bf16* __restrict__ qb, const __bf16* __restrict__ kb,
    const __bf16* __restrict__ vTb, const int* __restrict__ mask,
    __bf16* __restrict__ attn) {

    constexpr int KLD = 72;  // 64 elem row + 8 pad (TDM: interval 4, amount 3)
    constexpr int VLD = 40;  // 32 elem row + 8 pad (TDM: interval 3, amount 3)
    constexpr int PLD = 40;
    __shared__ alignas(16) __bf16 k_lds[32 * KLD];
    __shared__ alignas(16) __bf16 v_lds[64 * VLD];
    __shared__ alignas(16) __bf16 p_lds[8 * 16 * PLD];
    __shared__ int msk_lds[32];

    const int tid  = threadIdx.x;
    const int lane = tid & 31;
    const int wave = tid >> 5;
    const int nqb  = T_ / 128;
    const int bh   = blockIdx.x / nqb;
    const int qblk = blockIdx.x - bh * nqb;
    const int bb   = bh / H_;
    const int hh   = bh - bb * H_;
    const long kvbase = (long)bh * T_ * D_;
    const int q0 = qblk * 128 + wave * 16;

    const int mrow   = lane & 15;
    const int akb    = (lane >> 4) * 16;
    const int bkb    = (lane >> 4) * 16;
    const int rowadd = (lane >> 4) * 8;

    const unsigned k_off = (unsigned)(uintptr_t)(void*)k_lds;
    const unsigned v_off = (unsigned)(uintptr_t)(void*)v_lds;

    // Q fragments for this wave's 16 queries (A-operand, two K=32 chunks)
    BF16Frag qf[2];
    {
        const __bf16* qrow = qb + kvbase + (long)(q0 + mrow) * D_;
        qf[0].h[0] = *(const v8bf*)(qrow + akb);
        qf[0].h[1] = *(const v8bf*)(qrow + akb + 8);
        qf[1].h[0] = *(const v8bf*)(qrow + 32 + akb);
        qf[1].h[1] = *(const v8bf*)(qrow + 32 + akb + 8);
    }

    v8f o[4] = {vzero8(), vzero8(), vzero8(), vzero8()};
    float mi[8], li[8];
#pragma unroll
    for (int r = 0; r < 8; ++r) { mi[r] = -INFINITY; li[r] = 0.f; }

    __bf16* pw = &p_lds[wave * 16 * PLD];

    for (int kt = 0; kt < T_ / 32; ++kt) {
        // TDM stages K (32 rows x 64) and V^T (64 rows x 32) tiles into LDS
        if (wave == 0) {
            tdm_load_tile_2d(k_off, kb + kvbase + (long)kt * 32 * D_,
                             64, 32, 64, 32, 64ull, 4, 3);
            tdm_load_tile_2d(v_off, vTb + (long)bh * D_ * T_ + kt * 32,
                             32, 64, 32, 64, (unsigned long long)T_, 3, 3);
            __builtin_amdgcn_s_wait_tensorcnt(0);
        }
        if (tid < 32) msk_lds[tid] = mask[bb * T_ + kt * 32 + tid];
        __syncthreads();

        // S = Q x K^T for 32 keys: two 16x16 tiles, K-depth 64 = 2 WMMAs each
        v8f sa = vzero8(), sb = vzero8();
        BF16Frag kf;
        kf.h[0] = *(const v8bf*)&k_lds[mrow * KLD + bkb];
        kf.h[1] = *(const v8bf*)&k_lds[mrow * KLD + bkb + 8];
        sa = wmma_bf16(qf[0], kf, sa);
        kf.h[0] = *(const v8bf*)&k_lds[mrow * KLD + 32 + bkb];
        kf.h[1] = *(const v8bf*)&k_lds[mrow * KLD + 32 + bkb + 8];
        sa = wmma_bf16(qf[1], kf, sa);
        kf.h[0] = *(const v8bf*)&k_lds[(16 + mrow) * KLD + bkb];
        kf.h[1] = *(const v8bf*)&k_lds[(16 + mrow) * KLD + bkb + 8];
        sb = wmma_bf16(qf[0], kf, sb);
        kf.h[0] = *(const v8bf*)&k_lds[(16 + mrow) * KLD + 32 + bkb];
        kf.h[1] = *(const v8bf*)&k_lds[(16 + mrow) * KLD + 32 + bkb + 8];
        sb = wmma_bf16(qf[1], kf, sb);

        // key mask (lane's column = one key for all 8 rows)
        if (msk_lds[mrow] == 0) {
#pragma unroll
            for (int r = 0; r < 8; ++r) sa[r] = NEG_INF_;
        }
        if (msk_lds[16 + mrow] == 0) {
#pragma unroll
            for (int r = 0; r < 8; ++r) sb[r] = NEG_INF_;
        }

        // online softmax: row max/sum via butterfly within each 16-lane half
        float al[8];
#pragma unroll
        for (int r = 0; r < 8; ++r) {
            float v = fmaxf(sa[r], sb[r]);
#pragma unroll
            for (int off = 1; off < 16; off <<= 1) v = fmaxf(v, __shfl_xor(v, off, 32));
            float mn = fmaxf(mi[r], v);
            al[r] = __expf(mi[r] - mn);
            mi[r] = mn;
        }
#pragma unroll
        for (int r = 0; r < 8; ++r) {
            float pa = __expf(sa[r] - mi[r]);
            float pb = __expf(sb[r] - mi[r]);
            sa[r] = pa; sb[r] = pb;
            float s = pa + pb;
#pragma unroll
            for (int off = 1; off < 16; off <<= 1) s += __shfl_xor(s, off, 32);
            li[r] = li[r] * al[r] + s;
        }
#pragma unroll
        for (int j = 0; j < 4; ++j)
#pragma unroll
            for (int r = 0; r < 8; ++r) o[j][r] *= al[r];

        // restage P (C-layout, lane=col) -> LDS -> A-layout (lane=row)
#pragma unroll
        for (int r = 0; r < 8; ++r) {
            pw[(r + rowadd) * PLD + mrow]      = (__bf16)sa[r];
            pw[(r + rowadd) * PLD + 16 + mrow] = (__bf16)sb[r];
        }
        asm volatile("s_wait_dscnt 0" ::: "memory");  // wave-local RAW on LDS

        BF16Frag pf;
        pf.h[0] = *(const v8bf*)&pw[mrow * PLD + akb];
        pf.h[1] = *(const v8bf*)&pw[mrow * PLD + akb + 8];

        // O += P(16x32) x V(32x64): 4 d-tiles, V^T rows are B-operand rows
#pragma unroll
        for (int j = 0; j < 4; ++j) {
            BF16Frag vf;
            int dr = j * 16 + mrow;
            vf.h[0] = *(const v8bf*)&v_lds[dr * VLD + bkb];
            vf.h[1] = *(const v8bf*)&v_lds[dr * VLD + bkb + 8];
            o[j] = wmma_bf16(pf, vf, o[j]);
        }
        __syncthreads();  // protect LDS tiles before next TDM overwrite
    }

    // normalize + write attention output in [B*T, C] bf16 (c = h*64 + d)
#pragma unroll
    for (int j = 0; j < 4; ++j)
#pragma unroll
        for (int r = 0; r < 8; ++r) {
            int tt = q0 + r + rowadd;
            int dd = j * 16 + mrow;
            float val = o[j][r] / li[r];
            attn[((long)bb * T_ + tt) * C_ + hh * D_ + dd] = (__bf16)val;
        }
}

// ---------------- launcher ---------------------------------------------------
extern "C" void kernel_launch(void* const* d_in, const int* in_sizes, int n_in,
                              void* d_out, int out_size, void* d_ws, size_t ws_size,
                              hipStream_t stream) {
    (void)in_sizes; (void)n_in; (void)out_size; (void)ws_size;

    const float* x      = (const float*)d_in[0];
    const int*   mask   = (const int*)d_in[1];
    const float* w_qkv  = (const float*)d_in[2];
    const float* w_proj = (const float*)d_in[3];
    float*       out    = (float*)d_out;

    const long MT   = (long)B_ * T_;           // 8192 tokens
    const long NX   = MT * C_;                 // 8,388,608
    const long NQKV = (long)C_ * 3 * C_;       // 3,145,728
    const long NPRJ = (long)C_ * C_;           // 1,048,576
    const long NHED = (long)B_ * H_ * T_ * D_; // 8,388,608

    __bf16* xb     = (__bf16*)d_ws;
    __bf16* wqkvT  = xb + NX;
    __bf16* wprojT = wqkvT + NQKV;
    __bf16* qb     = wprojT + NPRJ;
    __bf16* kb     = qb + NHED;
    __bf16* vTb    = kb + NHED;
    __bf16* attnb  = vTb + NHED;

    convert_f32_bf16<<<(NX + 255) / 256, 256, 0, stream>>>(x, xb, (int)NX);
    transpose_to_bf16<<<(NQKV + 255) / 256, 256, 0, stream>>>(w_qkv, wqkvT, C_, 3 * C_);
    transpose_to_bf16<<<(NPRJ + 255) / 256, 256, 0, stream>>>(w_proj, wprojT, C_, C_);

    dim3 gqkv((unsigned)(MT / 128), (unsigned)(3 * C_ / 128));
    gemm_bf16_wmma<0><<<gqkv, 256, 0, stream>>>(xb, wqkvT, (int)MT, 3 * C_, C_,
                                                nullptr, qb, kb, vTb);

    flash_attn_wmma<<<B_ * H_ * (T_ / 128), 256, 0, stream>>>(qb, kb, vTb, mask, attnb);

    dim3 gprj((unsigned)(MT / 128), (unsigned)(C_ / 128));
    gemm_bf16_wmma<1><<<gprj, 256, 0, stream>>>(attnb, wprojT, (int)MT, C_, C_,
                                                out, nullptr, nullptr, nullptr);
}
